// BitLinearB158_46368466927646
// MI455X (gfx1250) — 8-bit matrix operand order repaired and checked by reading, compiled, not run on hardware
//
#include <hip/hip_runtime.h>
#include <hip/hip_bf16.h>


typedef __attribute__((ext_vector_type(8))) int v8i;

#define M_TOT 8192
#define N_TOT 11008
#define K_TOT 4096

#define BM 128
#define BN 64
#define BK 64

#define AUNIT 48
#define APL (BM * 2 * AUNIT)
#define BLDST 80
#define BPL (BN * BLDST)
#define BUFSZ (2 * APL + BPL)
#define SPER 9

#if defined(__has_builtin)
#if __has_builtin(__builtin_amdgcn_global_load_async_to_lds_b64) && \
    __has_builtin(__builtin_amdgcn_global_load_async_to_lds_b128) && \
    __has_builtin(__builtin_amdgcn_s_wait_asynccnt)
#define USE_ASYNC 1
#endif
#endif
#ifndef USE_ASYNC
#define USE_ASYNC 0
#endif

#if USE_ASYNC
typedef int avec2 __attribute__((__vector_size__(2 * sizeof(int))));
typedef int avec4 __attribute__((__vector_size__(4 * sizeof(int))));
#define GAS1 __attribute__((address_space(1)))
#define LAS3 __attribute__((address_space(3)))
#define ASYNC_WAIT(n) __builtin_amdgcn_s_wait_asynccnt(n)
#else
#define ASYNC_WAIT(n) ((void)0)
#endif

__device__ __forceinline__ void cp8_to_lds(signed char* s, const signed char* g) {
#if USE_ASYNC
  __builtin_amdgcn_global_load_async_to_lds_b64((GAS1 avec2*)g, (LAS3 avec2*)s, 0, 0);
#else
  *(int2*)s = *(const int2*)g;
#endif
}

__device__ __forceinline__ void cp16_to_lds(signed char* s, const signed char* g) {
#if USE_ASYNC
  __builtin_amdgcn_global_load_async_to_lds_b128((GAS1 avec4*)g, (LAS3 avec4*)s, 0, 0);
#else
  *(int4*)s = *(const int4*)g;
#endif
}

__global__ __launch_bounds__(256) void k_absmax(const float4* __restrict__ x,
                                                float* __restrict__ pmax, int n4) {
  __shared__ float red[256];
  float m = 0.0f;
  for (int i = blockIdx.x * blockDim.x + threadIdx.x; i < n4;
       i += gridDim.x * blockDim.x) {
    float4 v = x[i];
    m = fmaxf(m, fmaxf(fmaxf(fabsf(v.x), fabsf(v.y)),
                       fmaxf(fabsf(v.z), fabsf(v.w))));
  }
  red[threadIdx.x] = m;
  __syncthreads();
  for (int s = 128; s > 0; s >>= 1) {
    if (threadIdx.x < s) red[threadIdx.x] = fmaxf(red[threadIdx.x], red[threadIdx.x + s]);
    __syncthreads();
  }
  if (threadIdx.x == 0) pmax[blockIdx.x] = red[0];
}

__global__ __launch_bounds__(256) void k_abssum(const float4* __restrict__ w,
                                                float* __restrict__ psum, int n4) {
  __shared__ float red[256];
  float s = 0.0f;
  for (int i = blockIdx.x * blockDim.x + threadIdx.x; i < n4;
       i += gridDim.x * blockDim.x) {
    float4 v = w[i];
    s += fabsf(v.x) + fabsf(v.y) + fabsf(v.z) + fabsf(v.w);
  }
  red[threadIdx.x] = s;
  __syncthreads();
  for (int k = 128; k > 0; k >>= 1) {
    if (threadIdx.x < k) red[threadIdx.x] += red[threadIdx.x + k];
    __syncthreads();
  }
  if (threadIdx.x == 0) psum[blockIdx.x] = red[0];
}

__global__ __launch_bounds__(256) void k_finalize(const float* __restrict__ pmax,
                                                  const float* __restrict__ psum,
                                                  float* __restrict__ scal) {
  __shared__ float rm[256], rs[256];
  int t = threadIdx.x;
  float m = 0.0f, s = 0.0f;
  for (int i = t; i < 1024; i += 256) { m = fmaxf(m, pmax[i]); s += psum[i]; }
  rm[t] = m; rs[t] = s;
  __syncthreads();
  for (int k = 128; k > 0; k >>= 1) {
    if (t < k) { rm[t] = fmaxf(rm[t], rm[t + k]); rs[t] += rs[t + k]; }
    __syncthreads();
  }
  if (t == 0) {
    float gamma = fmaxf(rm[0], 1e-5f);
    float beta  = fmaxf(rs[0] * (1.0f / 45088768.0f), 1e-5f);
    scal[0] = 256.0f / gamma;
    scal[1] = 1.0f / beta;
    scal[2] = beta * gamma * (1.0f / 256.0f);
  }
}

__device__ __forceinline__ int qx1(float v, float qx) {
  float r = rintf(v * qx);
  r = fminf(fmaxf(r, -256.0f), 255.0f);
  return (int)r;
}

__global__ __launch_bounds__(256) void k_quant_x(const float4* __restrict__ x,
                                                 const float* __restrict__ scal,
                                                 char4* __restrict__ hi,
                                                 char4* __restrict__ lo) {
  const float qx = scal[0];
  int i = blockIdx.x * 256 + threadIdx.x;
  float4 v = x[i];
  int a = qx1(v.x, qx), b = qx1(v.y, qx), c = qx1(v.z, qx), d = qx1(v.w, qx);
  hi[i] = make_char4((signed char)(a >> 1), (signed char)(b >> 1),
                     (signed char)(c >> 1), (signed char)(d >> 1));
  lo[i] = make_char4((signed char)(a & 1), (signed char)(b & 1),
                     (signed char)(c & 1), (signed char)(d & 1));
}

__global__ __launch_bounds__(256) void k_quant_w(const float4* __restrict__ w,
                                                 const float* __restrict__ scal,
                                                 char4* __restrict__ wq) {
  const float ib = scal[1];
  int i = blockIdx.x * 256 + threadIdx.x;
  float4 v = w[i];
  signed char a = (signed char)fminf(rintf(fabsf(v.x) * ib), 1.0f);
  signed char b = (signed char)fminf(rintf(fabsf(v.y) * ib), 1.0f);
  signed char c = (signed char)fminf(rintf(fabsf(v.z) * ib), 1.0f);
  signed char d = (signed char)fminf(rintf(fabsf(v.w) * ib), 1.0f);
  wq[i] = make_char4(a, b, c, d);
}

__device__ __forceinline__ v8i frag_a(const signed char* p) {
  int4 u0 = *(const int4*)(p);
  int4 u1 = *(const int4*)(p + 16);
  v8i a;
  a[0] = u0.x; a[1] = u0.y; a[2] = u0.z; a[3] = u0.w;
  a[4] = u1.x; a[5] = u1.y; a[6] = u1.z; a[7] = u1.w;
  return a;
}

__device__ __forceinline__ v8i frag_b(const signed char* p) {
  int4 u0 = *(const int4*)(p);
  int4 u1 = *(const int4*)(p + 32);
  v8i b;
  b[0] = u0.x; b[1] = u0.y; b[2] = u0.z; b[3] = u0.w;
  b[4] = u1.x; b[5] = u1.y; b[6] = u1.z; b[7] = u1.w;
  return b;
}

__global__ __launch_bounds__(256) void k_gemm(const signed char* __restrict__ Ahi,
                                              const signed char* __restrict__ Alo,
                                              const signed char* __restrict__ Bq,
                                              const float* __restrict__ scal,
                                              float* __restrict__ out) {
  __shared__ __align__(16) signed char smem[2 * BUFSZ];

  const int tid  = threadIdx.x;
  const int lane = tid & 31;
  const int wave = tid >> 5;
  const int wm   = wave & 3;
  const int wn   = wave >> 2;
  const int half = lane >> 4;
  const int m16  = lane & 15;

  const int bm = blockIdx.y * BM;
  const int bn = blockIdx.x * BN;
  const float scale = scal[2];

  const signed char* AhiT = Ahi + (long)bm * K_TOT;
  const signed char* AloT = Alo + (long)bm * K_TOT;
  const signed char* BqT  = Bq  + (long)bn * K_TOT;

  int aGoff[4];
  int sAoffP[4];
  #pragma unroll
  for (int t = 0; t < 4; ++t) {
    const int c   = tid + 256 * t;
    const int row = c >> 3;
    const int q   = (c & 7) >> 1;
    const int h   = c & 1;
    aGoff[t]  = row * K_TOT + (q >> 1) * 32 + h * 16 + (q & 1) * 8;
    sAoffP[t] = (row * 2 + h) * AUNIT + q * 8;
  }
  const int brow  = tid >> 2;
  const int bseg  = (tid & 3) * 16;
  const int bGoff = brow * K_TOT + bseg;
  const int sBoff = brow * BLDST + bseg;

  const int aOff0 = ((wm * 32 +      m16) * 2 + half) * AUNIT;
  const int aOff1 = ((wm * 32 + 16 + m16) * 2 + half) * AUNIT;
  const int bOff0 = (wn * 32 +      m16) * BLDST + 16 * half;
  const int bOff1 = (wn * 32 + 16 + m16) * BLDST + 16 * half;

  v8i acc_hi[2][2] = {};
  v8i acc_lo[2][2] = {};

  signed char* buf0 = smem;
  signed char* buf1 = smem + BUFSZ;

  auto stage = [&](signed char* base, int kt) {
    const signed char* Ak = AhiT + kt * BK;
    const signed char* Lk = AloT + kt * BK;
    const signed char* Bk = BqT  + kt * BK;
    #pragma unroll
    for (int t = 0; t < 4; ++t) {
      cp8_to_lds(base + sAoffP[t],       Ak + aGoff[t]);
      cp8_to_lds(base + APL + sAoffP[t], Lk + aGoff[t]);
    }
    cp16_to_lds(base + 2 * APL + sBoff, Bk + bGoff);
  };

  auto compute = [&](const signed char* base) {
    const signed char* sAh = base;
    const signed char* sAl = base + APL;
    const signed char* sB  = base + 2 * APL;
    v8i b0  = frag_b(sB + bOff0);
    v8i b1  = frag_b(sB + bOff1);
    v8i ah0 = frag_a(sAh + aOff0);
    v8i ah1 = frag_a(sAh + aOff1);
    v8i al0 = frag_a(sAl + aOff0);
    v8i al1 = frag_a(sAl + aOff1);
    acc_hi[0][0] = __builtin_amdgcn_wmma_i32_16x16x64_iu8(true,  ah0, false, b0, acc_hi[0][0], false, false);
    acc_hi[0][1] = __builtin_amdgcn_wmma_i32_16x16x64_iu8(true,  ah0, false, b1, acc_hi[0][1], false, false);
    acc_hi[1][0] = __builtin_amdgcn_wmma_i32_16x16x64_iu8(true,  ah1, false, b0, acc_hi[1][0], false, false);
    acc_hi[1][1] = __builtin_amdgcn_wmma_i32_16x16x64_iu8(true,  ah1, false, b1, acc_hi[1][1], false, false);
    acc_lo[0][0] = __builtin_amdgcn_wmma_i32_16x16x64_iu8(false, al0, false, b0, acc_lo[0][0], false, false);
    acc_lo[0][1] = __builtin_amdgcn_wmma_i32_16x16x64_iu8(false, al0, false, b1, acc_lo[0][1], false, false);
    acc_lo[1][0] = __builtin_amdgcn_wmma_i32_16x16x64_iu8(false, al1, false, b0, acc_lo[1][0], false, false);
    acc_lo[1][1] = __builtin_amdgcn_wmma_i32_16x16x64_iu8(false, al1, false, b1, acc_lo[1][1], false, false);
  };

  const int KT = K_TOT / BK;

  stage(buf0, 0);
  for (int kt = 0; kt < KT; kt += 2) {
    stage(buf1, kt + 1);
    ASYNC_WAIT(SPER);
    __syncthreads();
    compute(buf0);
    __syncthreads();

    if (kt + 2 < KT) {
      stage(buf0, kt + 2);
      ASYNC_WAIT(SPER);
    } else {
      ASYNC_WAIT(0);
    }
    __syncthreads();
    compute(buf1);
    __syncthreads();
  }

  #pragma unroll
  for (int i = 0; i < 2; ++i) {
    #pragma unroll
    for (int j = 0; j < 2; ++j) {
      v8i r = (acc_hi[i][j] << 1) + acc_lo[i][j];
      const int gm = bm + wm * 32 + i * 16 + half * 8;
      const int gn = bn + wn * 32 + j * 16 + m16;
      float* o = out + (long)gm * N_TOT + gn;
      #pragma unroll
      for (int rr = 0; rr < 8; ++rr) {
        o[(long)rr * N_TOT] = (float)r[rr] * scale;
      }
    }
  }
}

extern "C" void kernel_launch(void* const* d_in, const int* in_sizes, int n_in,
                              void* d_out, int out_size, void* d_ws, size_t ws_size,
                              hipStream_t stream) {
  (void)in_sizes; (void)n_in; (void)out_size; (void)ws_size;
  const float* x = (const float*)d_in[0];
  const float* w = (const float*)d_in[1];
  float* out = (float*)d_out;

  char* ws = (char*)d_ws;
  float* pmax = (float*)(ws);
  float* psum = (float*)(ws + 4096);
  float* scal = (float*)(ws + 8192);
  signed char* Ahi = (signed char*)(ws + 16384);
  signed char* Alo = Ahi + (size_t)M_TOT * K_TOT;
  signed char* Wq  = Alo + (size_t)M_TOT * K_TOT;

  const int xn4 = M_TOT * K_TOT / 4;
  const int wn4 = N_TOT * K_TOT / 4;

  k_absmax<<<1024, 256, 0, stream>>>((const float4*)x, pmax, xn4);
  k_abssum<<<1024, 256, 0, stream>>>((const float4*)w, psum, wn4);
  k_finalize<<<1, 256, 0, stream>>>(pmax, psum, scal);
  k_quant_x<<<xn4 / 256, 256, 0, stream>>>((const float4*)x, scal,
                                           (char4*)Ahi, (char4*)Alo);
  k_quant_w<<<wn4 / 256, 256, 0, stream>>>((const float4*)w, scal, (char4*)Wq);

  dim3 grid(N_TOT / BN, M_TOT / BM);
  k_gemm<<<grid, 256, 0, stream>>>(Ahi, Alo, Wq, scal, out);
}
